// STBlock_85066122265300
// MI455X (gfx1250) — compile-verified, matrix operand order checked by reading, not run on hardware
//
#include <hip/hip_runtime.h>

// ---------------------------------------------------------------------------
// Constants from the reference
// ---------------------------------------------------------------------------
#define BB   64
#define TT   12
#define NN   1024
#define DIN  2
#define HID  64
#define ED   10
#define CP0  80    // layer0 concat channel 66 padded to 80  (2*CP0 = 160 = 5*32)
#define CP1  128   // layer1 concat channel 128              (2*CP1 = 256 = 8*32)

// ---------------------------------------------------------------------------
// bf16 helpers on raw bits (round-to-nearest-even)
// ---------------------------------------------------------------------------
__device__ __forceinline__ unsigned short f2bf(float f) {
    unsigned int u = __float_as_uint(f);
    unsigned int r = u + 0x7FFFu + ((u >> 16) & 1u);
    return (unsigned short)(r >> 16);
}
__device__ __forceinline__ float bf2f(unsigned short h) {
    return __uint_as_float(((unsigned int)h) << 16);
}

// ---------------------------------------------------------------------------
// WMMA types (wave32, 16x16x32 bf16 -> f32)
// ---------------------------------------------------------------------------
typedef __attribute__((ext_vector_type(16))) __bf16          v16bf;
typedef __attribute__((ext_vector_type(16))) unsigned short  v16us;
typedef __attribute__((ext_vector_type(8)))  float           v8f;

union BFVec { v16us u; v16bf b; };

// K-slot within a 16x32 (A) / 32x16 (B) bf16 fragment, per CDNA5 ISA layout:
// lanes 0-15 hold K {0..7,16..23}, lanes 16-31 hold K {8..15,24..31}.
// j = 0..15 walks two contiguous 8-element runs -> lowers to 2x b128 loads.
__device__ __forceinline__ int frag_k(int j, int hsel) {
    int v = j >> 1, hb = j & 1;
    return (v < 4) ? (hsel * 8 + 2 * v + hb)
                   : (16 + hsel * 8 + 2 * (v - 4) + hb);
}

// ---------------------------------------------------------------------------
// zero fill
// ---------------------------------------------------------------------------
__global__ void k_zero(float* p, int nElems) {
    int i = blockIdx.x * blockDim.x + threadIdx.x;
    if (i < nElems) p[i] = 0.0f;
}

// ---------------------------------------------------------------------------
// A = softmax(relu(E @ E^T), axis=1)  -> bf16 matrix + fp32 diagonal
// ---------------------------------------------------------------------------
__global__ void k_compute_A(const float* __restrict__ E,
                            unsigned short* __restrict__ Abf,
                            float* __restrict__ Adiag) {
    __shared__ float vals[NN];
    __shared__ float red[256];
    int n = blockIdx.x, tid = threadIdx.x;
    float e[ED];
#pragma unroll
    for (int i = 0; i < ED; ++i) e[i] = E[n * ED + i];

    for (int m = tid; m < NN; m += 256) {
        float s = 0.f;
#pragma unroll
        for (int i = 0; i < ED; ++i) s += e[i] * E[m * ED + i];
        vals[m] = s > 0.f ? s : 0.f;
    }
    __syncthreads();

    float mx = -1e30f;
    for (int m = tid; m < NN; m += 256) mx = fmaxf(mx, vals[m]);
    red[tid] = mx; __syncthreads();
    for (int s = 128; s > 0; s >>= 1) {
        if (tid < s) red[tid] = fmaxf(red[tid], red[tid + s]);
        __syncthreads();
    }
    mx = red[0]; __syncthreads();

    float sum = 0.f;
    for (int m = tid; m < NN; m += 256) {
        float ex = __expf(vals[m] - mx);
        vals[m] = ex; sum += ex;
    }
    red[tid] = sum; __syncthreads();
    for (int s = 128; s > 0; s >>= 1) {
        if (tid < s) red[tid] += red[tid + s];
        __syncthreads();
    }
    float inv = 1.0f / red[0];

    for (int m = tid; m < NN; m += 256) {
        float a = vals[m] * inv;
        Abf[(size_t)n * NN + m] = f2bf(a);
        if (m == n) Adiag[n] = a;
    }
}

// ---------------------------------------------------------------------------
// Per-node weights, K-contiguous transposed layout:
//   WnT[n, o, j] = sum_e E[n,e] * Wp[e, j/Cpad, j%Cpad, o]   (bf16, pad=0)
//   bn[n,o]     = sum_e E[n,e] * bp[e,o]
// ---------------------------------------------------------------------------
__global__ void k_nodeweights(const float* __restrict__ Wp,   // [ED,2,I,O]
                              const float* __restrict__ bp,   // [ED,O]
                              const float* __restrict__ E,
                              unsigned short* __restrict__ WnT,// [N, O, 2*Cpad]
                              float* __restrict__ bn,          // [N, O]
                              int I, int O, int Cpad) {
    int n = blockIdx.x, tid = threadIdx.x;
    float e[ED];
#pragma unroll
    for (int i = 0; i < ED; ++i) e[i] = E[n * ED + i];

    int K2 = 2 * Cpad;
    int total = K2 * O;
    unsigned short* Wnn = WnT + (size_t)n * K2 * O;
    for (int idx = tid; idx < total; idx += 256) {
        int o = idx / K2, j = idx - o * K2;      // j contiguous across tid
        int k = j / Cpad, i = j - k * Cpad;
        float v = 0.f;
        if (i < I) {
#pragma unroll
            for (int eidx = 0; eidx < ED; ++eidx)
                v += e[eidx] * Wp[((size_t)(eidx * 2 + k) * I + i) * O + o];
        }
        Wnn[idx] = f2bf(v);
    }
    for (int o = tid; o < O; o += 256) {
        float v = 0.f;
#pragma unroll
        for (int eidx = 0; eidx < ED; ++eidx) v += e[eidx] * bp[eidx * O + o];
        bn[(size_t)n * O + o] = v;
    }
}

// ---------------------------------------------------------------------------
// Concat builders (bf16), dual layout:
//   XcJ[b][n][j]   (A-operand of node GEMM: j contiguous)
//   XcT[b][j][n]   (B-operand of A@X GEMM: node contiguous)
// ---------------------------------------------------------------------------
__global__ void k_concat_l0(const float* __restrict__ x,   // [B,T,N,2]
                            const float* __restrict__ h,   // [B,N,64]
                            const float* __restrict__ zr,  // [B,N,128] (sigmoided)
                            int t, int use_z,
                            unsigned short* __restrict__ XcJ,
                            unsigned short* __restrict__ XcT) {
    int idx = blockIdx.x * 256 + threadIdx.x;
    const int total = BB * NN * CP0;
    if (idx >= total) return;
    int j = idx % CP0;
    int n = (idx / CP0) & (NN - 1);
    int b = idx / (CP0 * NN);
    float v = 0.f;
    if (j < DIN) {
        v = x[(((size_t)b * TT + t) * NN + n) * DIN + j];
    } else if (j < DIN + HID) {
        size_t bn = (size_t)b * NN + n;
        v = h[bn * HID + (j - DIN)];
        if (use_z) v *= zr[bn * (2 * HID) + (j - DIN)];
    }
    unsigned short bfv = f2bf(v);
    XcJ[idx] = bfv;
    XcT[((size_t)b * CP0 + j) * NN + n] = bfv;
}

__global__ void k_concat_l1(const unsigned short* __restrict__ buf0, // [B,T,N,64] bf16
                            const float* __restrict__ h,
                            const float* __restrict__ zr,
                            int t, int use_z,
                            unsigned short* __restrict__ XcJ,
                            unsigned short* __restrict__ XcT) {
    int idx = blockIdx.x * 256 + threadIdx.x;
    const int total = BB * NN * CP1;
    if (idx >= total) return;
    int j = idx % CP1;
    int n = (idx / CP1) & (NN - 1);
    int b = idx / (CP1 * NN);
    unsigned short bfv;
    if (j < HID) {
        bfv = buf0[(((size_t)b * TT + t) * NN + n) * HID + j];
    } else {
        size_t bn = (size_t)b * NN + n;
        float v = h[bn * HID + (j - HID)];
        if (use_z) v *= zr[bn * (2 * HID) + (j - HID)];
        bfv = f2bf(v);
    }
    XcJ[idx] = bfv;
    XcT[((size_t)b * CP1 + j) * NN + n] = bfv;
}

// ---------------------------------------------------------------------------
// Y[b] = A @ X[b]     A:[N,N] bf16, XcT:[B,Cpad,N] bf16 -> Y:[B,N,Cpad] bf16
// grid (N/64, Cpad/16, B): one wave computes 64x16 (4 M-tiles), the X column
// fragment is loaded once per K-step and reused across the 4 WMMAs.
// ---------------------------------------------------------------------------
__global__ __launch_bounds__(32)
void k_gemm_AX(const unsigned short* __restrict__ A,
               const unsigned short* __restrict__ XcT,
               unsigned short* __restrict__ Y, int Cpad) {
    int lane  = threadIdx.x;
    int mBase = blockIdx.x * 64;
    int cBase = blockIdx.y * 16;
    int b     = blockIdx.z;

    int m = lane & 15, hsel = lane >> 4, ncol = lane & 15;
    const unsigned short* Xrow = XcT + ((size_t)b * Cpad + cBase + ncol) * NN;

    v8f acc0 = {}, acc1 = {}, acc2 = {}, acc3 = {};
    for (int k0 = 0; k0 < NN; k0 += 32) {
        BFVec bv;
#pragma unroll
        for (int j = 0; j < 16; ++j) bv.u[j] = Xrow[k0 + frag_k(j, hsel)];

        BFVec a0, a1, a2, a3;
#pragma unroll
        for (int j = 0; j < 16; ++j) {
            int kk = k0 + frag_k(j, hsel);
            a0.u[j] = A[(size_t)(mBase + 0  + m) * NN + kk];
            a1.u[j] = A[(size_t)(mBase + 16 + m) * NN + kk];
            a2.u[j] = A[(size_t)(mBase + 32 + m) * NN + kk];
            a3.u[j] = A[(size_t)(mBase + 48 + m) * NN + kk];
        }
        acc0 = __builtin_amdgcn_wmma_f32_16x16x32_bf16(false, a0.b, false, bv.b, (short)0, acc0, false, false);
        acc1 = __builtin_amdgcn_wmma_f32_16x16x32_bf16(false, a1.b, false, bv.b, (short)0, acc1, false, false);
        acc2 = __builtin_amdgcn_wmma_f32_16x16x32_bf16(false, a2.b, false, bv.b, (short)0, acc2, false, false);
        acc3 = __builtin_amdgcn_wmma_f32_16x16x32_bf16(false, a3.b, false, bv.b, (short)0, acc3, false, false);
    }
    unsigned short* Yb = Y + (size_t)b * NN * Cpad + cBase + ncol;
#pragma unroll
    for (int i = 0; i < 8; ++i) {
        int mm = hsel * 8 + i;
        Yb[(size_t)(mBase + 0  + mm) * Cpad] = f2bf(acc0[i]);
        Yb[(size_t)(mBase + 16 + mm) * Cpad] = f2bf(acc1[i]);
        Yb[(size_t)(mBase + 32 + mm) * Cpad] = f2bf(acc2[i]);
        Yb[(size_t)(mBase + 48 + mm) * Cpad] = f2bf(acc3[i]);
    }
}

// ---------------------------------------------------------------------------
// Per-node einsum: out[b,n,o] = act( sum_j Xg[b,n,j]*W(n)[j,o] + bn[n,o] )
// Xg[j<Cpad] = XcJ (identity support), Xg[j>=Cpad] = AX (graph support)
// grid (N, O/16): one wave covers all 4 batch M-tiles (M=B=64), reusing the
// weight fragment across the 4 WMMAs per K-step.
// act: 0 = sigmoid (z|r gate), 1 = tanh (candidate)
// ---------------------------------------------------------------------------
__global__ __launch_bounds__(32)
void k_gemm_node(const unsigned short* __restrict__ XcJ,  // [B,N,Cpad]
                 const unsigned short* __restrict__ AX,   // [B,N,Cpad]
                 const unsigned short* __restrict__ WnT,  // [N,O,2*Cpad]
                 const float* __restrict__ bn,
                 float* __restrict__ out, int Cpad, int O, int act) {
    int lane  = threadIdx.x;
    int n     = blockIdx.x;
    int oBase = blockIdx.y * 16;
    int K2 = 2 * Cpad;

    int m = lane & 15, hsel = lane >> 4, ncol = lane & 15;
    const unsigned short* Wrow = WnT + ((size_t)n * O + oBase + ncol) * K2;
    size_t SB = (size_t)NN * Cpad;
    const unsigned short* Xn  = XcJ + (size_t)n * Cpad;
    const unsigned short* AXn = AX  + (size_t)n * Cpad;

    v8f acc0 = {}, acc1 = {}, acc2 = {}, acc3 = {};
    for (int k0 = 0; k0 < K2; k0 += 32) {
        BFVec bv;
#pragma unroll
        for (int j = 0; j < 16; ++j) bv.u[j] = Wrow[k0 + frag_k(j, hsel)];

        BFVec a0, a1, a2, a3;
#pragma unroll
        for (int j = 0; j < 16; ++j) {
            int jj = k0 + frag_k(j, hsel);
            const unsigned short* src = (jj < Cpad) ? Xn : (AXn - Cpad);
            a0.u[j] = src[(size_t)(0  + m) * SB + jj];
            a1.u[j] = src[(size_t)(16 + m) * SB + jj];
            a2.u[j] = src[(size_t)(32 + m) * SB + jj];
            a3.u[j] = src[(size_t)(48 + m) * SB + jj];
        }
        acc0 = __builtin_amdgcn_wmma_f32_16x16x32_bf16(false, a0.b, false, bv.b, (short)0, acc0, false, false);
        acc1 = __builtin_amdgcn_wmma_f32_16x16x32_bf16(false, a1.b, false, bv.b, (short)0, acc1, false, false);
        acc2 = __builtin_amdgcn_wmma_f32_16x16x32_bf16(false, a2.b, false, bv.b, (short)0, acc2, false, false);
        acc3 = __builtin_amdgcn_wmma_f32_16x16x32_bf16(false, a3.b, false, bv.b, (short)0, acc3, false, false);
    }

    int o = oBase + ncol;
    float bias = bn[(size_t)n * O + o];
#pragma unroll
    for (int i = 0; i < 8; ++i) {
        int mm = hsel * 8 + i;
        float v0 = acc0[i] + bias, v1 = acc1[i] + bias,
              v2 = acc2[i] + bias, v3 = acc3[i] + bias;
        if (act == 0) {
            v0 = 1.0f / (1.0f + __expf(-v0)); v1 = 1.0f / (1.0f + __expf(-v1));
            v2 = 1.0f / (1.0f + __expf(-v2)); v3 = 1.0f / (1.0f + __expf(-v3));
        } else {
            v0 = tanhf(v0); v1 = tanhf(v1); v2 = tanhf(v2); v3 = tanhf(v3);
        }
        out[((size_t)(0  + mm) * NN + n) * O + o] = v0;
        out[((size_t)(16 + mm) * NN + n) * O + o] = v1;
        out[((size_t)(32 + mm) * NN + n) * O + o] = v2;
        out[((size_t)(48 + mm) * NN + n) * O + o] = v3;
    }
}

// ---------------------------------------------------------------------------
// GRU combine: h = r*h + (1-r)*hc ; also emit bf16 timestep buffer
// ---------------------------------------------------------------------------
__global__ void k_gru_out(const float* __restrict__ zr,
                          const float* __restrict__ hc,
                          float* __restrict__ h,
                          unsigned short* __restrict__ buf, int t) {
    int idx = blockIdx.x * 256 + threadIdx.x;
    if (idx >= BB * NN * HID) return;
    int c = idx & (HID - 1);
    int n = (idx >> 6) & (NN - 1);
    int b = idx >> 16;
    size_t bn = (size_t)b * NN + n;
    float r  = zr[bn * (2 * HID) + HID + c];
    float hv = h[bn * HID + c];
    float nh = r * hv + (1.0f - r) * hc[bn * HID + c];
    h[bn * HID + c] = nh;
    buf[(((size_t)b * TT + t) * NN + n) * HID + c] = f2bf(nh);
}

// ---------------------------------------------------------------------------
// Fused tail: relation -> diag trick -> conv(Wc) -> cat(end_conv1) ->
// conv(W2)+b2 -> dilated conv(W3)+b3.  One block per (b,n), 64 threads.
// ---------------------------------------------------------------------------
__global__ __launch_bounds__(64)
void k_tail(const float* __restrict__ x,            // [B,T,N,2]
            const unsigned short* __restrict__ cur, // [B,T,N,64] bf16
            const float* __restrict__ Adiag,        // [N]
            const float* __restrict__ Wr,  const float* __restrict__ br,
            const float* __restrict__ W1,
            const float* __restrict__ Wc,           // [12,12,1,65]
            const float* __restrict__ W2,  const float* __restrict__ b2,
            const float* __restrict__ W3,  const float* __restrict__ b3,
            float* __restrict__ out) {
    __shared__ float rg2[TT][2 * HID];   // 12 x 128
    __shared__ float cat[TT][2 * HID];   // [rst | end_conv1]
    __shared__ float o2 [TT][HID + 1];   // 12 x 65

    int blk = blockIdx.x;
    int n = blk & (NN - 1);
    int b = blk / NN;
    int tid = threadIdx.x;   // 0..63
    float ad = Adiag[n];

    // relation + diagonal expansion, and end_conv1 (cur @ W1^T)
    for (int t = 0; t < TT; ++t) {
        size_t base = ((size_t)b * TT + t) * NN + n;
        float x0 = x[base * DIN + 0];
        float x1 = x[base * DIN + 1];
        float rel = x0 * Wr[tid * DIN + 0] + x1 * Wr[tid * DIN + 1] + br[tid];
        rg2[t][2 * tid + 0] = rel;        // k=0: diag(I)=1
        rg2[t][2 * tid + 1] = rel * ad;   // k=1: diag(A)
        float s = 0.f;
        const unsigned short* cv = cur + base * HID;
#pragma unroll
        for (int c2 = 0; c2 < HID; ++c2) s += bf2f(cv[c2]) * W1[tid * HID + c2];
        cat[t][HID + tid] = s;
    }
    __syncthreads();

    // rst = conv(rg2, Wc): out width 64, kernel (1,65) over 12 in-channels
    for (int idx = tid; idx < TT * HID; idx += 64) {
        int t = idx / HID, w = idx - t * HID;
        float s = 0.f;
        for (int i = 0; i < TT; ++i) {
            const float* wrow = Wc + (size_t)(t * TT + i) * (HID + 1);
            const float* grow = &rg2[i][w];
#pragma unroll 5
            for (int kw = 0; kw <= HID; ++kw) s += grow[kw] * wrow[kw];
        }
        cat[t][w] = s;
    }
    __syncthreads();

    // o2 = conv(cat, W2) + b2: out width 65, kernel (1,64)
    for (int idx = tid; idx < TT * (HID + 1); idx += 64) {
        int t = idx / (HID + 1), w = idx - t * (HID + 1);
        float s = b2[t];
        for (int i = 0; i < TT; ++i) {
            const float* wrow = W2 + (size_t)(t * TT + i) * HID;
            const float* crow = &cat[i][w];
#pragma unroll 4
            for (int kw = 0; kw < HID; ++kw) s += crow[kw] * wrow[kw];
        }
        o2[t][w] = s;
    }
    __syncthreads();

    // out3: dilated (1,2) kernel, dilation HID -> taps at w=0 and w=64
    if (tid < TT) {
        int t = tid;
        float s = b3[t];
#pragma unroll
        for (int i = 0; i < TT; ++i)
            s += o2[i][0] * W3[(t * TT + i) * 2 + 0]
               + o2[i][HID] * W3[(t * TT + i) * 2 + 1];
        out[((size_t)b * TT + t) * NN + n] = s;
    }
}

// ---------------------------------------------------------------------------
// Host orchestration
// ---------------------------------------------------------------------------
extern "C" void kernel_launch(void* const* d_in, const int* in_sizes, int n_in,
                              void* d_out, int out_size, void* d_ws, size_t ws_size,
                              hipStream_t stream) {
    const float* x   = (const float*)d_in[0];
    const float* E   = (const float*)d_in[1];
    const float* Wg0 = (const float*)d_in[2];
    const float* bg0 = (const float*)d_in[3];
    const float* Wu0 = (const float*)d_in[4];
    const float* bu0 = (const float*)d_in[5];
    const float* Wg1 = (const float*)d_in[6];
    const float* bg1 = (const float*)d_in[7];
    const float* Wu1 = (const float*)d_in[8];
    const float* bu1 = (const float*)d_in[9];
    const float* Wr  = (const float*)d_in[10];
    const float* br  = (const float*)d_in[11];
    const float* W1  = (const float*)d_in[12];
    const float* Wc  = (const float*)d_in[13];
    const float* W2  = (const float*)d_in[14];
    const float* b2  = (const float*)d_in[15];
    const float* W3  = (const float*)d_in[16];
    const float* b3  = (const float*)d_in[17];
    float* outp = (float*)d_out;

    // ---- workspace carve-out ----
    char* p = (char*)d_ws;
    auto take = [&](size_t bytes) { char* r = p; p += (bytes + 255) & ~(size_t)255; return r; };
    unsigned short* Abf   = (unsigned short*)take((size_t)NN * NN * 2);
    float*          Adiag = (float*)take((size_t)NN * 4);
    unsigned short* Wn_g0 = (unsigned short*)take((size_t)NN * 2 * CP0 * 128 * 2);
    float*          bn_g0 = (float*)take((size_t)NN * 128 * 4);
    unsigned short* Wn_u0 = (unsigned short*)take((size_t)NN * 2 * CP0 * 64 * 2);
    float*          bn_u0 = (float*)take((size_t)NN * 64 * 4);
    unsigned short* Wn_g1 = (unsigned short*)take((size_t)NN * 2 * CP1 * 128 * 2);
    float*          bn_g1 = (float*)take((size_t)NN * 128 * 4);
    unsigned short* Wn_u1 = (unsigned short*)take((size_t)NN * 2 * CP1 * 64 * 2);
    float*          bn_u1 = (float*)take((size_t)NN * 64 * 4);
    unsigned short* XcJ   = (unsigned short*)take((size_t)BB * NN * CP1 * 2);
    unsigned short* XcT   = (unsigned short*)take((size_t)BB * NN * CP1 * 2);
    unsigned short* AX    = (unsigned short*)take((size_t)BB * NN * CP1 * 2);
    float*          zr    = (float*)take((size_t)BB * NN * 128 * 4);
    float*          hc    = (float*)take((size_t)BB * NN * 64 * 4);
    float*          h     = (float*)take((size_t)BB * NN * 64 * 4);
    unsigned short* buf0  = (unsigned short*)take((size_t)BB * TT * NN * HID * 2);
    unsigned short* buf1  = (unsigned short*)take((size_t)BB * TT * NN * HID * 2);
    (void)ws_size; (void)in_sizes; (void)n_in; (void)out_size;

    // ---- adjacency + per-node weights (once; reused across T and B) ----
    k_compute_A<<<NN, 256, 0, stream>>>(E, Abf, Adiag);
    k_nodeweights<<<NN, 256, 0, stream>>>(Wg0, bg0, E, Wn_g0, bn_g0, DIN + HID, 128, CP0);
    k_nodeweights<<<NN, 256, 0, stream>>>(Wu0, bu0, E, Wn_u0, bn_u0, DIN + HID, 64,  CP0);
    k_nodeweights<<<NN, 256, 0, stream>>>(Wg1, bg1, E, Wn_g1, bn_g1, 2 * HID,   128, CP1);
    k_nodeweights<<<NN, 256, 0, stream>>>(Wu1, bu1, E, Wn_u1, bn_u1, 2 * HID,   64,  CP1);

    const int hElems = BB * NN * HID;
    dim3 gAX0(NN / 64, CP0 / 16, BB), gAX1(NN / 64, CP1 / 16, BB);
    dim3 gNg(NN, 128 / 16), gNu(NN, 64 / 16);
    int gC0 = (BB * NN * CP0 + 255) / 256;
    int gC1 = (BB * NN * CP1 + 255) / 256;
    int gH  = (hElems + 255) / 256;

    // ---- layer 0 (GRU over 12 timesteps) ----
    k_zero<<<gH, 256, 0, stream>>>(h, hElems);
    for (int t = 0; t < TT; ++t) {
        k_concat_l0<<<gC0, 256, 0, stream>>>(x, h, zr, t, 0, XcJ, XcT);
        k_gemm_AX<<<gAX0, 32, 0, stream>>>(Abf, XcT, AX, CP0);
        k_gemm_node<<<gNg, 32, 0, stream>>>(XcJ, AX, Wn_g0, bn_g0, zr, CP0, 128, 0);
        k_concat_l0<<<gC0, 256, 0, stream>>>(x, h, zr, t, 1, XcJ, XcT);
        k_gemm_AX<<<gAX0, 32, 0, stream>>>(Abf, XcT, AX, CP0);
        k_gemm_node<<<gNu, 32, 0, stream>>>(XcJ, AX, Wn_u0, bn_u0, hc, CP0, 64, 1);
        k_gru_out<<<gH, 256, 0, stream>>>(zr, hc, h, buf0, t);
    }

    // ---- layer 1 ----
    k_zero<<<gH, 256, 0, stream>>>(h, hElems);
    for (int t = 0; t < TT; ++t) {
        k_concat_l1<<<gC1, 256, 0, stream>>>(buf0, h, zr, t, 0, XcJ, XcT);
        k_gemm_AX<<<gAX1, 32, 0, stream>>>(Abf, XcT, AX, CP1);
        k_gemm_node<<<gNg, 32, 0, stream>>>(XcJ, AX, Wn_g1, bn_g1, zr, CP1, 128, 0);
        k_concat_l1<<<gC1, 256, 0, stream>>>(buf0, h, zr, t, 1, XcJ, XcT);
        k_gemm_AX<<<gAX1, 32, 0, stream>>>(Abf, XcT, AX, CP1);
        k_gemm_node<<<gNu, 32, 0, stream>>>(XcJ, AX, Wn_u1, bn_u1, hc, CP1, 64, 1);
        k_gru_out<<<gH, 256, 0, stream>>>(zr, hc, h, buf1, t);
    }

    // ---- fused conv tail ----
    k_tail<<<BB * NN, 64, 0, stream>>>(x, buf1, Adiag, Wr, br, W1, Wc, W2, b2, W3, b3, outp);
}
